// GoldenMoEPHFull_9981503995950
// MI455X (gfx1250) — matrix operand order repaired and checked by reading, not compiled, not run on hardware
//
#include <hip/hip_runtime.h>
#include <hip/hip_bf16.h>

// ---------------------------------------------------------------------------
// Problem constants (from reference): N=8192, D=1024, H=2048, O=1024, E=8
// ---------------------------------------------------------------------------
#define N_TOK 8192
#define DIM   1024
#define HID   2048
#define OUTD  1024
#define NEXP  8
#define TEMP_F 2.718281828459045f

typedef __attribute__((ext_vector_type(16))) __bf16 v16bf;
typedef __attribute__((ext_vector_type(8)))  float  v8f;
typedef __attribute__((ext_vector_type(4)))  int    v4i;

// pointee types for the async-DMA builtin (AS1 = global, AS3 = LDS)
typedef __attribute__((address_space(1))) v4i g_v4i;
typedef __attribute__((address_space(3))) v4i l_v4i;

#if defined(__has_builtin)
#if __has_builtin(__builtin_amdgcn_global_load_async_to_lds_b128)
#define HAVE_ASYNC_LDS 1
#endif
#endif

__device__ __forceinline__ void wait_async0() {
#if defined(__has_builtin) && __has_builtin(__builtin_amdgcn_s_wait_asynccnt)
  __builtin_amdgcn_s_wait_asynccnt(0);
#else
  asm volatile("s_wait_asynccnt 0x0" ::: "memory");
#endif
}

// f32 -> bf16 round-to-nearest-even, using __bf16 as storage only.
__device__ __forceinline__ __bf16 f2bf(float f) {
  unsigned u = __float_as_uint(f);
  unsigned r = (u + 0x7FFFu + ((u >> 16) & 1u)) >> 16;
  unsigned short s = (unsigned short)r;
  return *reinterpret_cast<__bf16*>(&s);
}

// ---------------------------------------------------------------------------
// 0) zero the small scalar accumulators in workspace
// ---------------------------------------------------------------------------
__global__ void init_scal_kernel(float* scal, int* kint) {
  if (threadIdx.x < 32) scal[threadIdx.x] = 0.0f;
  if (threadIdx.x == 0) *kint = 2;
}

// ---------------------------------------------------------------------------
// 1) per-row: gate softmax probs, ph signature match -> effective, clarity
// ---------------------------------------------------------------------------
__global__ __launch_bounds__(256) void row_features_kernel(
    const float* __restrict__ x,
    const float* __restrict__ gate_w, const float* __restrict__ gate_b,
    const float* __restrict__ ph_w1,  const float* __restrict__ ph_b1,
    const float* __restrict__ ph_w2,  const float* __restrict__ ph_b2,
    const float* __restrict__ cl_w1,  const float* __restrict__ cl_b1,
    const float* __restrict__ cl_w2,  const float* __restrict__ cl_b2,
    const float* __restrict__ sigs,
    float* __restrict__ probs, float* __restrict__ effective,
    float* __restrict__ scal)
{
  __shared__ float red[56][257];
  const int row = blockIdx.x;
  const int t = threadIdx.x;

  float accg[8], accp[32], accc[16];
#pragma unroll
  for (int i = 0; i < 8; ++i) accg[i] = 0.f;
#pragma unroll
  for (int i = 0; i < 32; ++i) accp[i] = 0.f;
#pragma unroll
  for (int i = 0; i < 16; ++i) accc[i] = 0.f;

  const float* xr = x + (size_t)row * DIM;
  for (int d = t; d < DIM; d += 256) {
    float xv = xr[d];
#pragma unroll
    for (int e = 0; e < 8; ++e)  accg[e] += xv * gate_w[d * 8 + e];
#pragma unroll
    for (int j = 0; j < 32; ++j) accp[j] += xv * ph_w1[d * 32 + j];
#pragma unroll
    for (int j = 0; j < 16; ++j) accc[j] += xv * cl_w1[d * 16 + j];
  }
#pragma unroll
  for (int e = 0; e < 8; ++e)  red[e][t]      = accg[e];
#pragma unroll
  for (int j = 0; j < 32; ++j) red[8 + j][t]  = accp[j];
#pragma unroll
  for (int j = 0; j < 16; ++j) red[40 + j][t] = accc[j];
  __syncthreads();

  if (t < 56) {
    float s = 0.f;
    for (int i = 0; i < 256; ++i) s += red[t][i];
    red[t][256] = s;
  }
  __syncthreads();

  if (t == 0) {
    float sc[8], pr[8];
#pragma unroll
    for (int e = 0; e < 8; ++e) sc[e] = (red[e][256] + gate_b[e]) * (1.0f / TEMP_F);
    float mx = sc[0];
#pragma unroll
    for (int e = 1; e < 8; ++e) mx = fmaxf(mx, sc[e]);
    float se = 0.f;
#pragma unroll
    for (int e = 0; e < 8; ++e) { pr[e] = __expf(sc[e] - mx); se += pr[e]; }
    float inv = 1.0f / se;
#pragma unroll
    for (int e = 0; e < 8; ++e) pr[e] *= inv;

    float t1[32];
#pragma unroll
    for (int j = 0; j < 32; ++j) {
      float v = red[8 + j][256] + ph_b1[j];
      t1[j] = v > 0.f ? v : 0.f;
    }
    float f2[16];
#pragma unroll
    for (int i = 0; i < 16; ++i) {
      float s = ph_b2[i];
      for (int j = 0; j < 32; ++j) s += t1[j] * ph_w2[j * 16 + i];
      f2[i] = s;
    }
    float nrm = 0.f;
#pragma unroll
    for (int i = 0; i < 16; ++i) nrm += f2[i] * f2[i];
    nrm = fmaxf(sqrtf(nrm), 1e-12f);
    float invn = 1.0f / nrm;
#pragma unroll
    for (int i = 0; i < 16; ++i) f2[i] *= invn;

    float cz = cl_b2[0];
#pragma unroll
    for (int j = 0; j < 16; ++j) {
      float v = red[40 + j][256] + cl_b1[j];
      v = v > 0.f ? v : 0.f;
      cz += v * cl_w2[j];
    }
    float clarity = 1.0f / (1.0f + __expf(-cz));
    atomicAdd(&scal[0], 8.0f - clarity * 6.0f);

    for (int e = 0; e < 8; ++e) {
      float sn = 0.f;
      for (int i = 0; i < 16; ++i) { float sv = sigs[e * 16 + i]; sn += sv * sv; }
      float isn = 1.0f / fmaxf(sqrtf(sn), 1e-12f);
      float dp = 0.f;
      for (int i = 0; i < 16; ++i) dp += f2[i] * sigs[e * 16 + i] * isn;
      float match = (dp + 1.0f) * 0.5f;
      probs[row * 8 + e] = pr[e];
      atomicAdd(&scal[1 + e], pr[e]);
      effective[row * 8 + e] = pr[e] * match;
    }
  }
}

// ---------------------------------------------------------------------------
// 2) k = clip(floor(mean(n_active)+0.5), 2, 8)
// ---------------------------------------------------------------------------
__global__ void compute_k_kernel(const float* __restrict__ scal, int* __restrict__ kint) {
  int k = (int)floorf(scal[0] * (1.0f / (float)N_TOK) + 0.5f);
  k = k < 2 ? 2 : (k > NEXP ? NEXP : k);
  *kint = k;
}

// ---------------------------------------------------------------------------
// 3) per-row top-k mask, renormalized weights, f counts
// ---------------------------------------------------------------------------
__global__ __launch_bounds__(256) void topk_weights_kernel(
    const float* __restrict__ effective, const int* __restrict__ kint,
    float* __restrict__ weights, float* __restrict__ scal)
{
  int row = blockIdx.x * blockDim.x + threadIdx.x;
  if (row >= N_TOK) return;
  int k = *kint;
  float eff[8], w[8];
#pragma unroll
  for (int i = 0; i < 8; ++i) eff[i] = effective[row * 8 + i];
  float s = 0.f;
#pragma unroll
  for (int i = 0; i < 8; ++i) {
    int rank = 0;
#pragma unroll
    for (int j = 0; j < 8; ++j)
      rank += (eff[j] > eff[i]) || (eff[j] == eff[i] && j < i);
    w[i] = (rank < k) ? eff[i] : 0.f;
    s += w[i];
  }
  float inv = 1.0f / (s + 1e-8f);
#pragma unroll
  for (int i = 0; i < 8; ++i) {
    float wv = w[i] * inv;
    weights[row * 8 + i] = wv;
    if (wv > 0.f) atomicAdd(&scal[9 + i], 1.0f);
  }
}

// ---------------------------------------------------------------------------
// 4) lb_loss -> d_out[N*O]
// ---------------------------------------------------------------------------
__global__ void lb_loss_kernel(const float* __restrict__ scal, float* __restrict__ out) {
  float lb = 0.f;
  for (int e = 0; e < 8; ++e)
    lb += (scal[9 + e] * (1.0f / (float)N_TOK)) * (scal[1 + e] * (1.0f / (float)N_TOK));
  out[(size_t)N_TOK * OUTD] = 0.01f * (float)NEXP * lb;
}

// ---------------------------------------------------------------------------
// 5) cast x -> bf16
// ---------------------------------------------------------------------------
__global__ __launch_bounds__(256) void cast_x_kernel(const float* __restrict__ x,
                                                     __bf16* __restrict__ xb) {
  size_t idx = (size_t)blockIdx.x * 256 + threadIdx.x;
  xb[idx] = f2bf(x[idx]);
}

// ---------------------------------------------------------------------------
// 5b) tiled transpose + cast: src f32 [K][Nn] -> dst bf16 [Nn][K]
// ---------------------------------------------------------------------------
__global__ __launch_bounds__(256) void transpose_cast_kernel(
    const float* __restrict__ src, __bf16* __restrict__ dst, int K, int Nn)
{
  __shared__ float tile[32][33];
  const int n0 = blockIdx.x * 32;
  const int k0 = blockIdx.y * 32;
  const int tx = threadIdx.x & 31;
  const int ty = threadIdx.x >> 5;
#pragma unroll
  for (int i = 0; i < 32; i += 8)
    tile[ty + i][tx] = src[(size_t)(k0 + ty + i) * Nn + n0 + tx];
  __syncthreads();
#pragma unroll
  for (int i = 0; i < 32; i += 8)
    dst[(size_t)(n0 + ty + i) * K + k0 + tx] = f2bf(tile[tx][ty + i]);
}

// ---------------------------------------------------------------------------
// 6) y init: y[n,o] = sum_e weights[n,e] * e_b2[e,o]
// ---------------------------------------------------------------------------
__global__ __launch_bounds__(256) void bias_init_kernel(const float* __restrict__ weights,
                                                        const float* __restrict__ e_b2,
                                                        float* __restrict__ y) {
  size_t idx = (size_t)blockIdx.x * 256 + threadIdx.x;
  int n = (int)(idx >> 10);
  int o = (int)(idx & 1023);
  float s = 0.f;
#pragma unroll
  for (int e = 0; e < 8; ++e) s += weights[n * 8 + e] * e_b2[e * OUTD + o];
  y[idx] = s;
}

// ---------------------------------------------------------------------------
// WMMA GEMM: C[M,N] = A[M,K] * Bt[N,K]^T, all bf16 operands, f32 accum.
// Block 128x256, 8 waves (2x4) of 64x64, K-tile 32.
// Data movement: GLOBAL_LOAD_ASYNC_TO_LDS_B128 (ASYNCcnt) into double-buffered
// LDS — no staging VGPRs, DMA of tile t+1 overlaps the 16 WMMAs of tile t.
// MODE 1: bias+relu+gate-scale -> bf16 Hout.   MODE 2: y += acc.
// ---------------------------------------------------------------------------
#define BM 128
#define BN 256
#define KT 32
#define LDA 40   // KT + 8 pad (bf16 units) -> 80B row stride, 16B aligned
#define LDB 40

template <int MODE>
__global__ __launch_bounds__(256, 1) void gemm_kernel(
    const __bf16* __restrict__ A,   // [M][K]
    const __bf16* __restrict__ Bt,  // [N][K] (pre-transposed weights)
    const float* __restrict__ bias, // MODE 1
    const float* __restrict__ wts,  // MODE 1: gate weights [N_TOK][8]
    int expert, int K,
    __bf16* __restrict__ Hout,      // MODE 1
    float* __restrict__ y)          // MODE 2
{
  __shared__ __bf16 As[2][BM][LDA];
  __shared__ __bf16 Bs[2][BN][LDB];

  const int t = threadIdx.x;
  const int lane = t & 31;
  const int wave = t >> 5;
  const int wm = wave >> 2, wn = wave & 3;
  const int half = lane >> 4, l16 = lane & 15;
  const int m0 = blockIdx.y * BM;
  const int n0 = blockIdx.x * BN;

  const int arow = t >> 1, aseg = t & 1;

  v8f acc[4][4] = {};

#if defined(HAVE_ASYNC_LDS)
  auto issue_tile = [&](int buf, int k0) {
    // A tile: 128x32 bf16 = 8KB, 2 x b128 per thread
    const __bf16* gpA = A + (size_t)(m0 + arow) * K + k0 + aseg * 16;
    __bf16* lpA = &As[buf][arow][aseg * 16];
    __builtin_amdgcn_global_load_async_to_lds_b128(
        (g_v4i*)gpA, (l_v4i*)lpA, 0, 0);
    __builtin_amdgcn_global_load_async_to_lds_b128(
        (g_v4i*)(gpA + 8), (l_v4i*)(lpA + 8), 0, 0);
    // B tile: 256x32 bf16 = 16KB, 4 x b128 per thread
#pragma unroll
    for (int i = 0; i < 4; ++i) {
      int idx = i * 256 + t;
      int brow = idx >> 2, bseg = idx & 3;
      const __bf16* gpB = Bt + (size_t)(n0 + brow) * K + k0 + bseg * 8;
      __bf16* lpB = &Bs[buf][brow][bseg * 8];
      __builtin_amdgcn_global_load_async_to_lds_b128(
          (g_v4i*)gpB, (l_v4i*)lpB, 0, 0);
    }
  };
#else
  auto stage_tile = [&](int buf, int k0) {
    const uint4* ap = reinterpret_cast<const uint4*>(
        A + (size_t)(m0 + arow) * K + k0 + aseg * 16);
    uint4 ra0 = ap[0], ra1 = ap[1];
    uint4* ad = reinterpret_cast<uint4*>(&As[buf][arow][aseg * 16]);
    ad[0] = ra0; ad[1] = ra1;
#pragma unroll
    for (int i = 0; i < 4; ++i) {
      int idx = i * 256 + t;
      int brow = idx >> 2, bseg = idx & 3;
      uint4 rb = *reinterpret_cast<const uint4*>(
          Bt + (size_t)(n0 + brow) * K + k0 + bseg * 8);
      *reinterpret_cast<uint4*>(&Bs[buf][brow][bseg * 8]) = rb;
    }
  };
#endif

  auto compute_tile = [&](int buf) {
    v16bf af[4], bfv[4];
#pragma unroll
    for (int i = 0; i < 4; ++i) {
      union { v16bf v; uint4 q[2]; } u;
      const __bf16* p = &As[buf][wm * 64 + i * 16 + l16][0];
      u.q[0] = *reinterpret_cast<const uint4*>(p + half * 8);
      u.q[1] = *reinterpret_cast<const uint4*>(p + 16 + half * 8);
      af[i] = u.v;
    }
#pragma unroll
    for (int i = 0; i < 4; ++i) {
      union { v16bf v; uint4 q[2]; } u;
      const __bf16* p = &Bs[buf][wn * 64 + i * 16 + l16][0];
      u.q[0] = *reinterpret_cast<const uint4*>(p + half * 8);
      u.q[1] = *reinterpret_cast<const uint4*>(p + 16 + half * 8);
      bfv[i] = u.v;
    }
#pragma unroll
    for (int mi = 0; mi < 4; ++mi)
#pragma unroll
      for (int ni = 0; ni < 4; ++ni)
        acc[mi][ni] = __builtin_amdgcn_wmma_f32_16x16x32_bf16(
            false, af[mi], false, bfv[ni], (short)0, acc[mi][ni], false, false);
  };

  const int T = K / KT;
#if defined(HAVE_ASYNC_LDS)
  issue_tile(0, 0);
  wait_async0();
  __syncthreads();
  for (int tt = 0; tt < T; ++tt) {
    const int cur = tt & 1;
    if (tt + 1 < T) issue_tile(1 - cur, (tt + 1) * KT);  // DMA overlaps WMMAs
    compute_tile(cur);
    if (tt + 1 < T) wait_async0();
    __syncthreads();
  }
#else
  for (int tt = 0; tt < T; ++tt) {
    __syncthreads();
    stage_tile(0, tt * KT);
    __syncthreads();
    compute_tile(0);
  }
#endif

  // epilogue
#pragma unroll
  for (int mi = 0; mi < 4; ++mi) {
#pragma unroll
    for (int r = 0; r < 8; ++r) {
      const int grow = m0 + wm * 64 + mi * 16 + half * 8 + r;
      if (MODE == 1) {
        const float rw = wts[grow * 8 + expert];
#pragma unroll
        for (int ni = 0; ni < 4; ++ni) {
          const int gcol = n0 + wn * 64 + ni * 16 + l16;
          float v = acc[mi][ni][r] + bias[gcol];
          v = v > 0.f ? v : 0.f;
          Hout[(size_t)grow * HID + gcol] = f2bf(v * rw);
        }
      } else {
#pragma unroll
        for (int ni = 0; ni < 4; ++ni) {
          const int gcol = n0 + wn * 64 + ni * 16 + l16;
          y[(size_t)grow * OUTD + gcol] += acc[mi][ni][r];
        }
      }
    }
  }
}

// ---------------------------------------------------------------------------
// launch
// ---------------------------------------------------------------------------
extern "C" void kernel_launch(void* const* d_in, const int* in_sizes, int n_in,
                              void* d_out, int out_size, void* d_ws, size_t ws_size,
                              hipStream_t stream) {
  const float* x      = (const float*)d_in[0];
  const float* gate_w = (const float*)d_in[1];
  const float* gate_b = (const float*)d_in[2];
  const float* ph_w1  = (const float*)d_in[3];
  const float* ph_b1  = (const float*)d_in[4];
  const float* ph_w2  = (const float*)d_in[5];
  const float* ph_b2  = (const float*)d_in[6];
  const float* cl_w1  = (const float*)d_in[7];
  const float* cl_b1  = (const float*)d_in[8];
  const float* cl_w2  = (const float*)d_in[9];
  const float* cl_b2  = (const float*)d_in[10];
  const float* sigs   = (const float*)d_in[11];
  const float* e_w1   = (const float*)d_in[12];
  const float* e_b1   = (const float*)d_in[13];
  const float* e_w2   = (const float*)d_in[14];
  const float* e_b2   = (const float*)d_in[15];
  float* y = (float*)d_out;

  char* W = (char*)d_ws;
  __bf16* xb       = (__bf16*)W;                                    // 16 MiB
  __bf16* hsb      = (__bf16*)(W + (size_t)16 * 1024 * 1024);       // 32 MiB
  __bf16* wtb      = (__bf16*)(W + (size_t)48 * 1024 * 1024);       // 4 MiB (W1t/W2t)
  float* probs     = (float*)(W + (size_t)52 * 1024 * 1024);
  float* effective = probs + (size_t)N_TOK * 8;
  float* weights   = effective + (size_t)N_TOK * 8;
  float* scal      = weights + (size_t)N_TOK * 8;                   // 32 floats
  int*   kint      = (int*)(scal + 32);

  init_scal_kernel<<<1, 64, 0, stream>>>(scal, kint);
  row_features_kernel<<<N_TOK, 256, 0, stream>>>(
      x, gate_w, gate_b, ph_w1, ph_b1, ph_w2, ph_b2,
      cl_w1, cl_b1, cl_w2, cl_b2, sigs, probs, effective, scal);
  compute_k_kernel<<<1, 1, 0, stream>>>(scal, kint);
  topk_weights_kernel<<<N_TOK / 256, 256, 0, stream>>>(effective, kint, weights, scal);
  lb_loss_kernel<<<1, 1, 0, stream>>>(scal, y);

  cast_x_kernel<<<(N_TOK * DIM) / 256, 256, 0, stream>>>(x, xb);
  bias_init_kernel<<<(N_TOK * OUTD) / 256, 256, 0, stream>>>(weights, e_b2, y);

  for (int e = 0; e < NEXP; ++e) {
    // W1_e [D][H] -> W1t [H][D] bf16
    transpose_cast_kernel<<<dim3(HID / 32, DIM / 32), 256, 0, stream>>>(
        e_w1 + (size_t)e * DIM * HID, wtb, DIM, HID);
    gemm_kernel<1><<<dim3(HID / BN, N_TOK / BM), 256, 0, stream>>>(
        xb, wtb, e_b1 + (size_t)e * HID, weights, e, DIM, hsb, nullptr);
    // W2_e [H][O] -> W2t [O][H] bf16
    transpose_cast_kernel<<<dim3(OUTD / 32, HID / 32), 256, 0, stream>>>(
        e_w2 + (size_t)e * HID * OUTD, wtb, HID, OUTD);
    gemm_kernel<2><<<dim3(OUTD / BN, N_TOK / BM), 256, 0, stream>>>(
        hsb, wtb, nullptr, nullptr, 0, HID, nullptr, y);
  }
}
